// GraphAttentionLayer_37984690765967
// MI455X (gfx1250) — matrix operand order repaired and checked by reading, not compiled, not run on hardware
//
#include <hip/hip_runtime.h>

typedef __attribute__((ext_vector_type(16))) _Float16 v16h;
typedef __attribute__((ext_vector_type(8)))  float    v8f;
typedef __attribute__((ext_vector_type(4)))  int      v4i;
typedef __attribute__((ext_vector_type(4)))  float    v4f;

#define GAT_ALPHA 0.2f
#define GAT_NEG_INF -9.0e15f
#define GAT_B 8
#define GAT_N 2048
#define GAT_F 64

// ---------------------------------------------------------------------------
// Kernel A: h = x @ W  (f32 accum) -> h16 (f16), s1 = h.a1, s2 = h.a2
// grid = B*N blocks, 64 threads (one output feature each).
// W (16KB) and a stay L2/L0 resident; real traffic is x (4MB) + h16 (2MB).
// ---------------------------------------------------------------------------
__global__ __launch_bounds__(64)
void gat_proj(const float* __restrict__ x, const float* __restrict__ W,
              const float* __restrict__ a, _Float16* __restrict__ h16,
              float* __restrict__ s1, float* __restrict__ s2)
{
    __shared__ float xr[GAT_F];
    __shared__ float r1[GAT_F];
    __shared__ float r2[GAT_F];
    const int g = blockIdx.x;          // flat (b, n) row index
    const int t = threadIdx.x;         // output feature
    xr[t] = x[(size_t)g * GAT_F + t];
    __syncthreads();
    float acc = 0.f;
    #pragma unroll
    for (int k = 0; k < GAT_F; ++k)
        acc = fmaf(xr[k], W[k * GAT_F + t], acc);
    h16[(size_t)g * GAT_F + t] = (_Float16)acc;
    r1[t] = acc * a[t];
    r2[t] = acc * a[GAT_F + t];
    __syncthreads();
    #pragma unroll
    for (int s = 32; s > 0; s >>= 1) {
        if (t < s) { r1[t] += r1[t + s]; r2[t] += r2[t + s]; }
        __syncthreads();
    }
    if (t == 0) { s1[g] = r1[0]; s2[g] = r2[0]; }
}

// ---------------------------------------------------------------------------
// Kernel B: fused masked-attention + softmax + (attn @ h) + elu.
// One wave32 per 16-row query tile; online softmax; A = f16 probs,
// B = f16 h rows, C/D = f32 via v_wmma_f32_16x16x32_f16 (4 col tiles).
// ---------------------------------------------------------------------------
__global__ __launch_bounds__(32)
void gat_attn(const int* __restrict__ adj, const float* __restrict__ s1g,
              const float* __restrict__ s2g, const _Float16* __restrict__ h16,
              float* __restrict__ out)
{
    const int tile = blockIdx.x;                 // 0 .. B*(N/16)-1
    const int b    = tile >> 7;                  // / (N/16) = /128
    const int i0   = (tile & 127) << 4;          // query tile start
    const int lane = threadIdx.x;
    const int r    = lane & 15;                  // my A-fragment row (M)
    const int hi   = lane >> 4;                  // K half selector
    const int row  = i0 + r;

    const int*      adjrow = adj + ((size_t)b * GAT_N + row) * GAT_N;
    const float     s1r    = s1g[b * GAT_N + row];
    const float*    s2b    = s2g + b * GAT_N;
    const _Float16* hb     = h16 + (size_t)b * GAT_N * GAT_F;

    float m = -__builtin_inff();
    float l = 0.f;
    v8f acc0 = {}, acc1 = {}, acc2 = {}, acc3 = {};

    for (int j0 = 0; j0 < GAT_N; j0 += 32) {
        const int kA = j0 + hi * 8;              // K-slots 0..7 (this lane)
        const int kB = kA + 16;                  // K-slots 8..15 (this lane)

        // adj streamed exactly once -> nontemporal; prefetch next tile.
        v4i a0 = __builtin_nontemporal_load((const v4i*)(adjrow + kA));
        v4i a1 = __builtin_nontemporal_load((const v4i*)(adjrow + kA + 4));
        v4i a2 = __builtin_nontemporal_load((const v4i*)(adjrow + kB));
        v4i a3 = __builtin_nontemporal_load((const v4i*)(adjrow + kB + 4));
        __builtin_prefetch(adjrow + kA + 32, 0, 1);
        v4f t0 = *(const v4f*)(s2b + kA);
        v4f t1 = *(const v4f*)(s2b + kA + 4);
        v4f t2 = *(const v4f*)(s2b + kB);
        v4f t3 = *(const v4f*)(s2b + kB + 4);

        float ev[16];
        #pragma unroll
        for (int q = 0; q < 4; ++q) {
            float e;
            e = s1r + t0[q]; e = e > 0.f ? e : GAT_ALPHA * e; ev[q]      = a0[q] > 0 ? e : GAT_NEG_INF;
            e = s1r + t1[q]; e = e > 0.f ? e : GAT_ALPHA * e; ev[4 + q]  = a1[q] > 0 ? e : GAT_NEG_INF;
            e = s1r + t2[q]; e = e > 0.f ? e : GAT_ALPHA * e; ev[8 + q]  = a2[q] > 0 ? e : GAT_NEG_INF;
            e = s1r + t3[q]; e = e > 0.f ? e : GAT_ALPHA * e; ev[12 + q] = a3[q] > 0 ? e : GAT_NEG_INF;
        }

        // online softmax: row stats duplicated in lanes r and r+16
        float tmax = ev[0];
        #pragma unroll
        for (int q = 1; q < 16; ++q) tmax = fmaxf(tmax, ev[q]);
        tmax = fmaxf(tmax, __shfl_xor(tmax, 16, 32));
        const float mnew  = fmaxf(m, tmax);
        const float scale = __expf(m - mnew);    // first iter: exp(-inf)=0

        float psum = 0.f;
        v16h pa;
        #pragma unroll
        for (int q = 0; q < 16; ++q) {
            const float p = __expf(ev[q] - mnew);
            psum += p;
            pa[q] = (_Float16)p;                 // matches 16-bit A VGPR layout
        }
        psum += __shfl_xor(psum, 16, 32);
        l = l * scale + psum;
        m = mnew;

        if (__any(scale != 1.0f)) {              // rescale accumulators (rare)
            #pragma unroll
            for (int v = 0; v < 8; ++v) {        // C row = v + 8*hi
                const float sv = __shfl(scale, hi ? v + 8 : v, 32);
                acc0[v] *= sv; acc1[v] *= sv; acc2[v] *= sv; acc3[v] *= sv;
            }
        }

        // B fragment: lane l holds column N=l%16, elements over K in table order (matches A).
        v16h b0, b1, b2, b3;
        #pragma unroll
        for (int q = 0; q < 16; ++q) {
            const _Float16* hrow = hb + (size_t)(j0 + hi * 8 + q + (q >= 8 ? 8 : 0)) * GAT_F + r;
            b0[q] = hrow[0]; b1[q] = hrow[16]; b2[q] = hrow[32]; b3[q] = hrow[48];
        }

        acc0 = __builtin_amdgcn_wmma_f32_16x16x32_f16(false, pa, false, b0, (short)0, acc0, false, false);
        acc1 = __builtin_amdgcn_wmma_f32_16x16x32_f16(false, pa, false, b1, (short)0, acc1, false, false);
        acc2 = __builtin_amdgcn_wmma_f32_16x16x32_f16(false, pa, false, b2, (short)0, acc2, false, false);
        acc3 = __builtin_amdgcn_wmma_f32_16x16x32_f16(false, pa, false, b3, (short)0, acc3, false, false);
    }

    // Epilogue: divide by row sum, elu, coalesced stores.
    #pragma unroll
    for (int v = 0; v < 8; ++v) {
        const float lv = __shfl(l, hi ? v + 8 : v, 32);
        const float rv = 1.0f / lv;              // lv > 0 always (uniform if fully masked)
        const size_t obase = ((size_t)b * GAT_N + i0 + v + hi * 8) * GAT_F + r;
        float z;
        z = acc0[v] * rv; out[obase +  0] = z > 0.f ? z : __expf(z) - 1.f;
        z = acc1[v] * rv; out[obase + 16] = z > 0.f ? z : __expf(z) - 1.f;
        z = acc2[v] * rv; out[obase + 32] = z > 0.f ? z : __expf(z) - 1.f;
        z = acc3[v] * rv; out[obase + 48] = z > 0.f ? z : __expf(z) - 1.f;
    }
}

// ---------------------------------------------------------------------------
extern "C" void kernel_launch(void* const* d_in, const int* in_sizes, int n_in,
                              void* d_out, int out_size, void* d_ws, size_t ws_size,
                              hipStream_t stream) {
    const float* x   = (const float*)d_in[0];   // (8,2048,64) f32
    const int*   adj = (const int*)  d_in[1];   // (8,2048,2048) i32
    const float* W   = (const float*)d_in[2];   // (64,64) f32
    const float* a   = (const float*)d_in[3];   // (128,1) f32
    float* out = (float*)d_out;                 // (8,2048,64) f32

    char* ws = (char*)d_ws;
    _Float16* h16 = (_Float16*)ws;                                   // 2 MB
    float*    s1  = (float*)(ws + (size_t)GAT_B * GAT_N * GAT_F * 2);
    float*    s2  = s1 + GAT_B * GAT_N;                              // +64 KB each

    gat_proj<<<GAT_B * GAT_N, 64, 0, stream>>>(x, W, a, h16, s1, s2);
    gat_attn<<<GAT_B * (GAT_N / 16), 32, 0, stream>>>(adj, s1, s2, h16, out);
}
